// MultiHeadAttention_30356828848776
// MI455X (gfx1250) — matrix operand order repaired and checked by reading, not compiled, not run on hardware
//
#include <hip/hip_runtime.h>
#include <hip/hip_bf16.h>
#include <stdint.h>

typedef unsigned short u16;
typedef __attribute__((ext_vector_type(16))) __bf16 v16bf;
typedef __attribute__((ext_vector_type(8)))  float  v8f;
typedef __attribute__((ext_vector_type(4)))  unsigned int u32x4;
typedef __attribute__((ext_vector_type(8)))  int i32x8;
typedef __attribute__((ext_vector_type(4)))  int i32x4;

#define B_    4
#define S_    2048
#define DIN_  1024
#define E_    1024
#define H_    16
#define HD_   64
#define MTOT  (B_ * S_)          // 8192
#define N_QKV (3 * E_)           // 3072

// ---------------- helpers ----------------

__device__ __forceinline__ u16 f2bf(float f) {
  unsigned u = __builtin_bit_cast(unsigned, f);
  u += 0x7FFFu + ((u >> 16) & 1u);          // round-to-nearest-even
  return (u16)(u >> 16);
}

// A-matrix fragment (16x32 bf16, MxK) from row-major LDS tile [rows][ld].
__device__ __forceinline__ v16bf lds_a_frag(const u16* lds, int row0, int ld,
                                            int k0, int lane) {
  const int half = (lane >> 4) & 1, m = lane & 15;
  union { v16bf v; unsigned u[8]; } f;
  const u16* rp = lds + (row0 + m) * ld + k0 + half * 16;
#pragma unroll
  for (int i = 0; i < 8; ++i) {
    const int k = 2 * i;
    f.u[i] = *reinterpret_cast<const unsigned*>(rp + k);
  }
  return f.v;
}

// B-matrix fragment (32x16 bf16, KxN) from K-major ("B^T", [N][ld]) LDS tile.
__device__ __forceinline__ v16bf lds_bt_frag(const u16* lds, int n0, int ld,
                                             int k0, int lane) {
  const int half = (lane >> 4) & 1, n = lane & 15;
  union { v16bf v; unsigned u[8]; } f;
  const u16* rp = lds + (n0 + n) * ld + k0 + half * 16;
#pragma unroll
  for (int i = 0; i < 8; ++i)
    f.u[i] = *reinterpret_cast<const unsigned*>(rp + 2 * i);
  return f.v;
}

// B-matrix fragment (32x16) gathered from row-major [K][ld] LDS tile.
__device__ __forceinline__ v16bf lds_b_gather(const u16* lds, int k0, int ld,
                                              int n0, int lane) {
  const int half = (lane >> 4) & 1, n = lane & 15;
  union { v16bf v; unsigned u[8]; } f;
  const u16* cp = lds + (k0 + 16 * half) * ld + n0 + n;
#pragma unroll
  for (int i = 0; i < 8; ++i) {
    unsigned lo = cp[(2 * i) * ld];
    unsigned hi = cp[(2 * i + 1) * ld];
    f.u[i] = lo | (hi << 16);
  }
  return f.v;
}

#define WMMA_BF16(a, b, c) \
  __builtin_amdgcn_wmma_f32_16x16x32_bf16(false, (a), false, (b), (short)0, (c), false, false)

__device__ __forceinline__ float rmax16(float x) {
#pragma unroll
  for (int off = 8; off > 0; off >>= 1) x = fmaxf(x, __shfl_xor(x, off, 16));
  return x;
}
__device__ __forceinline__ float rsum16(float x) {
#pragma unroll
  for (int off = 8; off > 0; off >>= 1) x += __shfl_xor(x, off, 16);
  return x;
}

// ---------------- Tensor Data Mover: 2D bf16 tile -> LDS ----------------
// D# group0: count=1 | lds_addr | 57b global addr | type=2
// D# group1: data_size=2B, pad_enable, tensor dims == tile dims (no OOB),
//            tensor_dim0_stride = row pitch (elements).
// pad_icode: pad after 2^pad_icode 8-byte units stored.  pad_acode: (code+1) DWORDs.
__device__ __forceinline__ void tdm_load_tile(unsigned lds_off, const u16* gptr,
                                              unsigned tile_w, unsigned tile_h,
                                              unsigned row_stride,
                                              unsigned pad_icode, unsigned pad_acode) {
  unsigned long long ga = (unsigned long long)(uintptr_t)gptr;
  u32x4 g0;
  g0[0] = 1u;                                            // count=1, user mode
  g0[1] = lds_off;                                       // lds_addr
  g0[2] = (unsigned)ga;                                  // global_addr[31:0]
  g0[3] = (unsigned)((ga >> 32) & 0x01FFFFFFu) | (2u << 30);  // addr[56:32] | type=2
  i32x8 g1;
  g1[0] = (int)((1u << 16) | (1u << 20) | (pad_icode << 22) | (pad_acode << 25));
  g1[1] = (int)((tile_w & 0xFFFFu) << 16);               // tensor_dim0 lo -> bits 63:48
  g1[2] = (int)(((tile_w >> 16) & 0xFFFFu) | ((tile_h & 0xFFFFu) << 16)); // d0 hi | d1 lo
  g1[3] = (int)(((tile_h >> 16) & 0xFFFFu) | (tile_w << 16));             // d1 hi | tile_dim0
  g1[4] = (int)(tile_h & 0xFFFFu);                       // tile_dim1 (tile_dim2 = 0)
  g1[5] = (int)row_stride;                               // tensor_dim0_stride lo32
  g1[6] = 0;
  g1[7] = 0;
  i32x4 z4 = {0, 0, 0, 0};
  i32x8 z8 = {0, 0, 0, 0, 0, 0, 0, 0};
  __builtin_amdgcn_tensor_load_to_lds(g0, g1, z4, z4, z8, 0);
}

__device__ __forceinline__ unsigned lds_off_of(const void* p) {
  return (unsigned)(uintptr_t)p;   // flat shared addr: low 32 bits = LDS offset
}

// ---------------- conversion kernels ----------------

__global__ void cvt_bf16(const float* __restrict__ src, u16* __restrict__ dst, int n) {
  int i = blockIdx.x * 256 + threadIdx.x;
  if (i < n) dst[i] = f2bf(src[i]);
}

// src row-major [K][N] -> dst [N][K] bf16
__global__ void transpose_cvt(const float* __restrict__ src, u16* __restrict__ dst,
                              int K, int N) {
  int i = blockIdx.x * 256 + threadIdx.x;
  if (i < K * N) {
    int n = i / K, k = i - n * K;
    dst[(size_t)n * K + k] = f2bf(src[(size_t)k * N + n]);
  }
}

// ---------------- QKV projection GEMM ----------------
#define LDK 40   // 32 K-values + 16B TDM row pad  (rows: 64B data + 16B pad)

__global__ __launch_bounds__(256) void gemm_qkv(const u16* __restrict__ xb,
                                                const u16* __restrict__ wt,
                                                const float* __restrict__ bias,
                                                u16* __restrict__ qb,
                                                u16* __restrict__ kb,
                                                u16* __restrict__ vb) {
  __shared__ alignas(16) u16 Xs[2][64 * LDK];
  __shared__ alignas(16) u16 Ws[2][128 * LDK];
  const int tid = threadIdx.x, lane = tid & 31, wid = tid >> 5;
  const int wm = wid >> 1, wn = wid & 1;
  const int m0 = blockIdx.x * 64, n0b = blockIdx.y * 128;
  const u16* xg = xb + (size_t)m0 * DIN_;
  const u16* wg = wt + (size_t)n0b * DIN_;

  if (wid == 0) {  // prime the pipeline: K-chunk 0 via TDM
    tdm_load_tile(lds_off_of(&Xs[0][0]), xg, 32, 64, DIN_, 3, 3);
    tdm_load_tile(lds_off_of(&Ws[0][0]), wg, 32, 128, DIN_, 3, 3);
  }

  v8f acc[4] = {};
#pragma unroll 1
  for (int kt = 0; kt < DIN_ / 32; ++kt) {
    if (wid == 0) __builtin_amdgcn_s_wait_tensorcnt(0);
    __syncthreads();                     // tile kt ready; tile kt-1 readers done
    if (wid == 0 && kt + 1 < DIN_ / 32) {  // overlap DMA of kt+1 with compute of kt
      tdm_load_tile(lds_off_of(&Xs[(kt + 1) & 1][0]), xg + (kt + 1) * 32, 32, 64, DIN_, 3, 3);
      tdm_load_tile(lds_off_of(&Ws[(kt + 1) & 1][0]), wg + (kt + 1) * 32, 32, 128, DIN_, 3, 3);
    }
    const u16* xt = Xs[kt & 1];
    const u16* wtile = Ws[kt & 1];
    v16bf a = lds_a_frag(xt, wm * 16, LDK, 0, lane);
#pragma unroll
    for (int t = 0; t < 4; ++t) {
      v16bf b = lds_bt_frag(wtile, wn * 64 + t * 16, LDK, 0, lane);
      acc[t] = WMMA_BF16(a, b, acc[t]);
    }
  }

  // epilogue: +bias, fold 1/sqrt(HD) into Q, scatter to [B][H][S][64] bf16
  const int half = lane >> 4, col = lane & 15;
#pragma unroll
  for (int t = 0; t < 4; ++t) {
    const int n = n0b + wn * 64 + t * 16 + col;
    const int h = n / 192, rr = n - h * 192;
    const int which = rr >> 6, d = rr & 63;
    u16* dst = (which == 0) ? qb : (which == 1) ? kb : vb;
    const float bsc = bias[n];
#pragma unroll
    for (int r = 0; r < 8; ++r) {
      const int mg = m0 + wm * 16 + r + 8 * half;
      const int batch = mg >> 11, s = mg & (S_ - 1);
      float v = acc[t][r] + bsc;
      if (which == 0) v *= 0.125f;  // 1/sqrt(64)
      dst[(((size_t)batch * H_ + h) * S_ + s) * HD_ + d] = f2bf(v);
    }
  }
}

// ---------------- flash attention ----------------
#define LDQ 72   // 64 elems (128B) + 16B TDM row pad
#define LDP 40

__global__ __launch_bounds__(256) void attn_fwd(const u16* __restrict__ qb,
                                                const u16* __restrict__ kb,
                                                const u16* __restrict__ vb,
                                                u16* __restrict__ vals) {
  __shared__ alignas(16) u16 Qs[128 * LDQ];
  __shared__ alignas(16) u16 Ks[2][32 * LDQ];
  __shared__ alignas(16) u16 Vs[2][32 * LDQ];
  __shared__ alignas(16) u16 Ps[8][16 * LDP];

  const int tid = threadIdx.x, lane = tid & 31, wid = tid >> 5;
  const int half = lane >> 4, col = lane & 15;
  const int blk = blockIdx.x;
  const int qchunk = blk & 15, bh = blk >> 4;
  const int batch = bh >> 4, h = bh & 15;
  const size_t bhoff = (size_t)bh * S_ * HD_;
  const u16* Qg = qb + bhoff;
  const u16* Kg = kb + bhoff;
  const u16* Vg = vb + bhoff;
  const int qs0 = qchunk * 128;

  if (wid == 0) {  // Q slab (once) + first K/V chunk, all via TDM
    tdm_load_tile(lds_off_of(&Qs[0]), Qg + (size_t)qs0 * HD_, 64, 128, HD_, 4, 3);
    tdm_load_tile(lds_off_of(&Ks[0][0]), Kg, 64, 32, HD_, 4, 3);
    tdm_load_tile(lds_off_of(&Vs[0][0]), Vg, 64, 32, HD_, 4, 3);
  }

  v16bf qa0, qa1;
  float m_r[8], l_r[8];
#pragma unroll
  for (int r = 0; r < 8; ++r) { m_r[r] = -1e30f; l_r[r] = 0.f; }
  v8f oc[4] = {};

#pragma unroll 1
  for (int kt = 0; kt < S_ / 32; ++kt) {
    if (wid == 0) __builtin_amdgcn_s_wait_tensorcnt(0);
    __syncthreads();                       // chunk kt (and Q on kt==0) ready
    if (wid == 0 && kt + 1 < S_ / 32) {    // prefetch next K/V chunk
      tdm_load_tile(lds_off_of(&Ks[(kt + 1) & 1][0]), Kg + (size_t)(kt + 1) * 32 * HD_,
                    64, 32, HD_, 4, 3);
      tdm_load_tile(lds_off_of(&Vs[(kt + 1) & 1][0]), Vg + (size_t)(kt + 1) * 32 * HD_,
                    64, 32, HD_, 4, 3);
    }
    if (kt == 0) {
      qa0 = lds_a_frag(Qs, wid * 16, LDQ, 0, lane);
      qa1 = lds_a_frag(Qs, wid * 16, LDQ, 32, lane);
    }
    const u16* kt_s = Ks[kt & 1];
    const u16* vt_s = Vs[kt & 1];

    // logits: two 16x16 tiles (keys +0..15, +16..31), contraction over HD=64
    v8f s0 = {}, s1 = {};
    s0 = WMMA_BF16(qa0, lds_bt_frag(kt_s, 0, LDQ, 0, lane), s0);
    s0 = WMMA_BF16(qa1, lds_bt_frag(kt_s, 0, LDQ, 32, lane), s0);
    s1 = WMMA_BF16(qa0, lds_bt_frag(kt_s, 16, LDQ, 0, lane), s1);
    s1 = WMMA_BF16(qa1, lds_bt_frag(kt_s, 16, LDQ, 32, lane), s1);

    // online softmax (row r + 8*half lives across one 16-lane group)
#pragma unroll
    for (int r = 0; r < 8; ++r) {
      float mx = rmax16(fmaxf(s0[r], s1[r]));
      float nm = fmaxf(m_r[r], mx);
      float p0 = __expf(s0[r] - nm);
      float p1 = __expf(s1[r] - nm);
      float alpha = __expf(m_r[r] - nm);
      m_r[r] = nm;
      l_r[r] = l_r[r] * alpha + rsum16(p0 + p1);
#pragma unroll
      for (int t = 0; t < 4; ++t) oc[t][r] *= alpha;
      s0[r] = p0; s1[r] = p1;
    }

    // reshape P (C-layout f32) -> A-layout bf16 via per-wave LDS buffer
    u16* pb = &Ps[wid][0];
#pragma unroll
    for (int r = 0; r < 8; ++r) {
      const int row = r + 8 * half;
      pb[row * LDP + col]      = f2bf(s0[r]);
      pb[row * LDP + 16 + col] = f2bf(s1[r]);
    }
    asm volatile("s_wait_dscnt 0x0" ::: "memory");  // same-wave LDS RAW
    v16bf pa = lds_a_frag(pb, 0, LDP, 0, lane);

#pragma unroll
    for (int t = 0; t < 4; ++t)
      oc[t] = WMMA_BF16(pa, lds_b_gather(vt_s, 0, LDQ, t * 16, lane), oc[t]);
  }

  // normalize and emit [B][S][E] bf16 (column = h*64 + d)
#pragma unroll
  for (int t = 0; t < 4; ++t) {
#pragma unroll
    for (int r = 0; r < 8; ++r) {
      const int srow = qs0 + wid * 16 + r + 8 * half;
      const int cg = h * HD_ + t * 16 + col;
      vals[((size_t)batch * S_ + srow) * E_ + cg] = f2bf(oc[t][r] / l_r[r]);
    }
  }
}

// ---------------- output projection GEMM (fp32 out) ----------------

__global__ __launch_bounds__(256) void gemm_out(const u16* __restrict__ ab,
                                                const u16* __restrict__ wt,
                                                const float* __restrict__ bias,
                                                float* __restrict__ out) {
  __shared__ alignas(16) u16 Xs[2][64 * LDK];
  __shared__ alignas(16) u16 Ws[2][128 * LDK];
  const int tid = threadIdx.x, lane = tid & 31, wid = tid >> 5;
  const int wm = wid >> 1, wn = wid & 1;
  const int m0 = blockIdx.x * 64, n0b = blockIdx.y * 128;
  const u16* xg = ab + (size_t)m0 * E_;
  const u16* wg = wt + (size_t)n0b * E_;

  if (wid == 0) {
    tdm_load_tile(lds_off_of(&Xs[0][0]), xg, 32, 64, E_, 3, 3);
    tdm_load_tile(lds_off_of(&Ws[0][0]), wg, 32, 128, E_, 3, 3);
  }

  v8f acc[4] = {};
#pragma unroll 1
  for (int kt = 0; kt < E_ / 32; ++kt) {
    if (wid == 0) __builtin_amdgcn_s_wait_tensorcnt(0);
    __syncthreads();
    if (wid == 0 && kt + 1 < E_ / 32) {
      tdm_load_tile(lds_off_of(&Xs[(kt + 1) & 1][0]), xg + (kt + 1) * 32, 32, 64, E_, 3, 3);
      tdm_load_tile(lds_off_of(&Ws[(kt + 1) & 1][0]), wg + (kt + 1) * 32, 32, 128, E_, 3, 3);
    }
    const u16* xt = Xs[kt & 1];
    const u16* wtile = Ws[kt & 1];
    v16bf a = lds_a_frag(xt, wm * 16, LDK, 0, lane);
#pragma unroll
    for (int t = 0; t < 4; ++t) {
      v16bf b = lds_bt_frag(wtile, wn * 64 + t * 16, LDK, 0, lane);
      acc[t] = WMMA_BF16(a, b, acc[t]);
    }
  }

  const int half = lane >> 4, col = lane & 15;
#pragma unroll
  for (int t = 0; t < 4; ++t) {
    const int n = n0b + wn * 64 + t * 16 + col;
    const float bsc = bias[n];
#pragma unroll
    for (int r = 0; r < 8; ++r) {
      const int mg = m0 + wm * 16 + r + 8 * half;
      out[(size_t)mg * E_ + n] = acc[t][r] + bsc;
    }
  }
}

// ---------------- host launch ----------------

extern "C" void kernel_launch(void* const* d_in, const int* in_sizes, int n_in,
                              void* d_out, int out_size, void* d_ws, size_t ws_size,
                              hipStream_t stream) {
  const float* x     = (const float*)d_in[0];
  const float* w_qkv = (const float*)d_in[1];
  const float* b_qkv = (const float*)d_in[2];
  const float* w_o   = (const float*)d_in[3];
  const float* b_o   = (const float*)d_in[4];
  float* out = (float*)d_out;

  char* ws = (char*)d_ws;
  size_t off = 0;
  u16* xb    = (u16*)(ws + off); off += (size_t)MTOT * DIN_ * 2;
  u16* wqkvT = (u16*)(ws + off); off += (size_t)N_QKV * DIN_ * 2;
  u16* woT   = (u16*)(ws + off); off += (size_t)E_ * E_ * 2;
  u16* qb    = (u16*)(ws + off); off += (size_t)B_ * H_ * S_ * HD_ * 2;
  u16* kb    = (u16*)(ws + off); off += (size_t)B_ * H_ * S_ * HD_ * 2;
  u16* vb    = (u16*)(ws + off); off += (size_t)B_ * H_ * S_ * HD_ * 2;
  u16* vals  = (u16*)(ws + off); off += (size_t)MTOT * E_ * 2;

  {
    int n = MTOT * DIN_;
    cvt_bf16<<<(n + 255) / 256, 256, 0, stream>>>(x, xb, n);
  }
  {
    int n = DIN_ * N_QKV;
    transpose_cvt<<<(n + 255) / 256, 256, 0, stream>>>(w_qkv, wqkvT, DIN_, N_QKV);
  }
  {
    int n = E_ * E_;
    transpose_cvt<<<(n + 255) / 256, 256, 0, stream>>>(w_o, woT, E_, E_);
  }

  gemm_qkv<<<dim3(MTOT / 64, N_QKV / 128), 256, 0, stream>>>(xb, wqkvT, b_qkv, qb, kb, vb);
  attn_fwd<<<B_ * H_ * (S_ / 128), 256, 0, stream>>>(qb, kb, vb, vals);
  gemm_out<<<dim3(MTOT / 64, E_ / 128), 256, 0, stream>>>(vals, woT, b_o, out);
}
